// CrossModalBlock_31207232373126
// MI455X (gfx1250) — matrix operand order repaired and checked by reading, not compiled, not run on hardware
//
#include <hip/hip_runtime.h>

#define B_ 16
#define P_ 576
#define N_ 512
#define D_ 1024
#define H_ 16
#define HID_ 4096

typedef __attribute__((ext_vector_type(16))) __bf16   bf16x16;
typedef __attribute__((ext_vector_type(8)))  float    f32x8;
typedef __attribute__((ext_vector_type(4)))  float    f32x4;
typedef __attribute__((ext_vector_type(4)))  unsigned u32x4;

union FragB { u32x4 u[2]; bf16x16 v; };

__device__ __forceinline__ unsigned short f2bf(float f) {
  unsigned u = __builtin_bit_cast(unsigned, f);
  unsigned r = u + 0x7FFFu + ((u >> 16) & 1u);
  return (unsigned short)(r >> 16);
}

__device__ __forceinline__ f32x8 zero8() {
  f32x8 z;
#pragma unroll
  for (int i = 0; i < 8; ++i) z[i] = 0.0f;
  return z;
}

__device__ __forceinline__ f32x8 wmma_bf16(const FragB& a, const FragB& b, f32x8 c) {
  return __builtin_amdgcn_wmma_f32_16x16x32_bf16(false, a.v, false, b.v,
                                                 (short)0, c, false, false);
}

// Flat shared-memory address low 32 bits == wave-relative LDS byte offset
// (ISA 10.2: LDS aperture maps addr[31:0] directly).
__device__ __forceinline__ unsigned lds_off(const void* p) {
  return (unsigned)(unsigned long long)p;
}
// ASYNCcnt-tracked DMA of 16B from global memory into LDS (GV mode).
__device__ __forceinline__ void async_ld_b128(unsigned lds_dst, const void* gsrc) {
  asm volatile("global_load_async_to_lds_b128 %0, %1, off"
               :: "v"(lds_dst), "v"(gsrc) : "memory");
}
__device__ __forceinline__ void wait_asynccnt0() {
  asm volatile("s_wait_asynccnt 0x0" ::: "memory");
}
// Async loads complete in order: <=4 outstanding => the older tile has landed.
__device__ __forceinline__ void wait_asynccnt4() {
  asm volatile("s_wait_asynccnt 0x4" ::: "memory");
}

// ---------------------------------------------------------------------------
// fp32 -> bf16 conversion (grid-stride)
// ---------------------------------------------------------------------------
__global__ void cvt_bf16_kernel(const float* __restrict__ in,
                                unsigned short* __restrict__ out, long n) {
  for (long i = (long)blockIdx.x * 256 + threadIdx.x; i < n; i += 256L * gridDim.x)
    out[i] = f2bf(in[i]);
}

// ---------------------------------------------------------------------------
// Vt[b,h,j,n] = kvt_bf16[b,n, D + h*64 + j]   (makes A·V a K-contiguous GEMM)
// ---------------------------------------------------------------------------
__global__ void vtrans_kernel(const unsigned short* __restrict__ kvt,
                              unsigned short* __restrict__ vt) {
  long idx = (long)blockIdx.x * 256 + threadIdx.x;       // B*H*64*512 total
  int  n = (int)(idx & 511);
  long r = idx >> 9;
  int  j = (int)(r & 63);  r >>= 6;
  int  h = (int)(r & 15);
  long b = r >> 4;
  vt[idx] = kvt[(b * N_ + n) * (2L * D_) + D_ + h * 64 + j];
}

// ---------------------------------------------------------------------------
// Generic bf16 GEMM: C[M,N] = A[M,K] * W[N,K]^T + bias  (tile 128x128x32)
// Double-buffered async global->LDS staging: tile kt+1 streams in under the
// 8 WMMAs of tile kt (ASYNCcnt in-order completion => s_wait_asynccnt 4).
// LDS rows padded to 40 elems (80B): 16 fragment lanes hit 16 distinct banks.
// ---------------------------------------------------------------------------
#define LSTR 40
__global__ __launch_bounds__(256)
void gemm_bf16_kernel(const unsigned short* __restrict__ A, int lda,
                      const unsigned short* __restrict__ W, int ldb,
                      const float* __restrict__ bias,
                      float* __restrict__ Cf, unsigned short* __restrict__ Cb,
                      int ldc, int K, int relu) {
  __shared__ __align__(16) unsigned short As[2][128 * LSTR];
  __shared__ __align__(16) unsigned short Bs[2][128 * LSTR];

  const int tid = threadIdx.x, lane = tid & 31, wave = tid >> 5;
  const int wm = (wave & 1) * 64;        // wave M offset within tile
  const int wn = (wave >> 1) * 32;       // wave N offset within tile
  const long m0 = (long)blockIdx.y * 128;
  const long n0 = (long)blockIdx.x * 128;

  const int  arow = tid >> 1, akh = (tid & 1) * 16;      // 16 bf16 per thread
  const unsigned short* Ag = A + (m0 + arow) * (long)lda + akh;
  const unsigned short* Bg = W + (n0 + arow) * (long)ldb + akh;

  f32x8 acc[4][2];
#pragma unroll
  for (int i = 0; i < 4; ++i)
#pragma unroll
    for (int j = 0; j < 2; ++j) acc[i][j] = zero8();

  const int nk = K / 32;

  const unsigned aoff[2] = { lds_off(&As[0][arow * LSTR + akh]),
                             lds_off(&As[1][arow * LSTR + akh]) };
  const unsigned boff[2] = { lds_off(&Bs[0][arow * LSTR + akh]),
                             lds_off(&Bs[1][arow * LSTR + akh]) };

  auto issue_tile = [&](int buf, int kt) {
    const unsigned short* Ap = Ag + (long)kt * 32;
    const unsigned short* Bp = Bg + (long)kt * 32;
    async_ld_b128(aoff[buf],      Ap);
    async_ld_b128(aoff[buf] + 16, Ap + 8);
    async_ld_b128(boff[buf],      Bp);
    async_ld_b128(boff[buf] + 16, Bp + 8);
  };

  issue_tile(0, 0);
  for (int kt = 0; kt < nk; ++kt) {
    if (kt + 1 < nk) {
      issue_tile((kt + 1) & 1, kt + 1);    // overlaps this tile's WMMAs
      if (kt + 2 < nk) {                   // gfx1250 global_prefetch_b8
        __builtin_prefetch(Ag + (kt + 2) * 32, 0, 1);
        __builtin_prefetch(Bg + (kt + 2) * 32, 0, 1);
      }
      wait_asynccnt4();                    // tile kt landed; kt+1 in flight
    } else {
      wait_asynccnt0();
    }
    __syncthreads();                       // all waves' tile-kt stores visible

    const unsigned short* Asb = As[kt & 1];
    const unsigned short* Bsb = Bs[kt & 1];

    FragB bfr[2], afr[4];
#pragma unroll
    for (int nt = 0; nt < 2; ++nt) {
      int n = wn + nt * 16 + (lane & 15);
      int kb = (lane >> 4) * 16;
      bfr[nt].u[0] = *(const u32x4*)&Bsb[n * LSTR + kb];
      bfr[nt].u[1] = *(const u32x4*)&Bsb[n * LSTR + kb + 8];
    }
#pragma unroll
    for (int mt = 0; mt < 4; ++mt) {
      int m = wm + mt * 16 + (lane & 15);
      int ka = (lane >> 4) * 16;
      afr[mt].u[0] = *(const u32x4*)&Asb[m * LSTR + ka];
      afr[mt].u[1] = *(const u32x4*)&Asb[m * LSTR + ka + 8];
    }
#pragma unroll
    for (int mt = 0; mt < 4; ++mt)
#pragma unroll
      for (int nt = 0; nt < 2; ++nt)
        acc[mt][nt] = wmma_bf16(afr[mt], bfr[nt], acc[mt][nt]);

    __syncthreads();   // reads of buf[kt&1] done before it is overwritten
  }

#pragma unroll
  for (int mt = 0; mt < 4; ++mt)
#pragma unroll
    for (int nt = 0; nt < 2; ++nt) {
      long n = n0 + wn + nt * 16 + (lane & 15);
      float bv = bias ? bias[n] : 0.0f;
#pragma unroll
      for (int g = 0; g < 8; ++g) {
        long m = m0 + wm + mt * 16 + g + ((lane >> 4) * 8);
        float v = acc[mt][nt][g] + bv;
        if (relu) v = fmaxf(v, 0.0f);
        if (Cf) Cf[m * (long)ldc + n] = v;
        if (Cb) Cb[m * (long)ldc + n] = f2bf(v);
      }
    }
}

// ---------------------------------------------------------------------------
// scores[b,h,p,n] = (Q_bh[p,:] . K_bh[n,:]) * exp(-log_tau[h]) / 8
// tile: M=64 (P), N=128 (text), K=64 fixed; grid (4, 9, B*H)
// Q/K tiles staged with async global->LDS (no VGPR round-trip).
// ---------------------------------------------------------------------------
__global__ __launch_bounds__(256)
void attn_scores_kernel(const unsigned short* __restrict__ qb,
                        const unsigned short* __restrict__ kvtb,
                        const float* __restrict__ log_tau,
                        float* __restrict__ S) {
  __shared__ __align__(16) unsigned short Qs[64 * 64];
  __shared__ __align__(16) unsigned short Ks[128 * 64];

  const int tid = threadIdx.x, lane = tid & 31, wave = tid >> 5;
  const int b = blockIdx.z >> 4, h = blockIdx.z & 15;
  const long p0 = (long)blockIdx.y * 64;
  const long n0 = (long)blockIdx.x * 128;
  const int wm = (wave & 1) * 32, wn = (wave >> 1) * 32;

  { // Q tile 64x64: 16 bf16/thread
    int r = tid >> 2, kh = (tid & 3) * 16;
    const unsigned short* g = qb + ((long)b * P_ + p0 + r) * D_ + h * 64 + kh;
    unsigned q = lds_off(&Qs[r * 64 + kh]);
    async_ld_b128(q,      g);
    async_ld_b128(q + 16, g + 8);
  }
  { // K tile 128x64: 32 bf16/thread
    int r = tid >> 1, kh = (tid & 1) * 32;
    const unsigned short* g = kvtb + ((long)b * N_ + n0 + r) * (2L * D_) + h * 64 + kh;
    unsigned k = lds_off(&Ks[r * 64 + kh]);
    async_ld_b128(k,      g);
    async_ld_b128(k + 16, g + 8);
    async_ld_b128(k + 32, g + 16);
    async_ld_b128(k + 48, g + 24);
  }
  wait_asynccnt0();
  __syncthreads();

  f32x8 acc[2][2];
#pragma unroll
  for (int i = 0; i < 2; ++i)
#pragma unroll
    for (int j = 0; j < 2; ++j) acc[i][j] = zero8();

#pragma unroll
  for (int ks = 0; ks < 64; ks += 32) {
    FragB bfr[2], afr[2];
#pragma unroll
    for (int nt = 0; nt < 2; ++nt) {
      int n = wn + nt * 16 + (lane & 15);
      int kb = ks + (lane >> 4) * 16;
      bfr[nt].u[0] = *(const u32x4*)&Ks[n * 64 + kb];
      bfr[nt].u[1] = *(const u32x4*)&Ks[n * 64 + kb + 8];
    }
#pragma unroll
    for (int mt = 0; mt < 2; ++mt) {
      int m = wm + mt * 16 + (lane & 15);
      int ka = ks + (lane >> 4) * 16;
      afr[mt].u[0] = *(const u32x4*)&Qs[m * 64 + ka];
      afr[mt].u[1] = *(const u32x4*)&Qs[m * 64 + ka + 8];
    }
#pragma unroll
    for (int mt = 0; mt < 2; ++mt)
#pragma unroll
      for (int nt = 0; nt < 2; ++nt)
        acc[mt][nt] = wmma_bf16(afr[mt], bfr[nt], acc[mt][nt]);
  }

  const float scale = 0.125f * __expf(-log_tau[h]);
  float* Sb = S + ((long)(b * H_ + h)) * P_ * N_;
#pragma unroll
  for (int mt = 0; mt < 2; ++mt)
#pragma unroll
    for (int nt = 0; nt < 2; ++nt)
#pragma unroll
      for (int g = 0; g < 8; ++g) {
        long p = p0 + wm + mt * 16 + g + ((lane >> 4) * 8);
        long n = n0 + wn + nt * 16 + (lane & 15);
        Sb[p * N_ + n] = acc[mt][nt][g] * scale;
      }
}

// ---------------------------------------------------------------------------
// row softmax over N=512 with mask, in place (one row per block)
// ---------------------------------------------------------------------------
__global__ __launch_bounds__(256)
void softmax_kernel(float* __restrict__ S, const unsigned char* __restrict__ mask) {
  __shared__ float red[256];
  const long row = blockIdx.x;                 // (b*H + h)*P + p
  const long b = row / (long)(H_ * P_);
  float* Sr = S + row * N_;
  const unsigned char* mr = mask + b * N_;
  const int tid = threadIdx.x;

  float v0 = mr[tid]       ? Sr[tid]       : -INFINITY;
  float v1 = mr[tid + 256] ? Sr[tid + 256] : -INFINITY;
  red[tid] = fmaxf(v0, v1);
  __syncthreads();
  for (int s = 128; s > 0; s >>= 1) {
    if (tid < s) red[tid] = fmaxf(red[tid], red[tid + s]);
    __syncthreads();
  }
  float mx = red[0];
  __syncthreads();
  float e0 = (v0 == -INFINITY) ? 0.0f : __expf(v0 - mx);
  float e1 = (v1 == -INFINITY) ? 0.0f : __expf(v1 - mx);
  red[tid] = e0 + e1;
  __syncthreads();
  for (int s = 128; s > 0; s >>= 1) {
    if (tid < s) red[tid] += red[tid + s];
    __syncthreads();
  }
  float inv = 1.0f / red[0];
  Sr[tid]       = e0 * inv;
  Sr[tid + 256] = e1 * inv;
}

// ---------------------------------------------------------------------------
// attn[b,p,h*64+j] = sum_n W[b,h,p,n] * V[b,n,h*64+j]
// A = fp32 weights (converted to bf16 while staging), B = Vt[b,h,64,512]
// V tile staged async; W tile needs VALU convert so it goes via ds_store.
// tile: M=64, N=64, K-loop 16x32; grid (1, 9, B*H)
// ---------------------------------------------------------------------------
__global__ __launch_bounds__(256)
void attn_av_kernel(const float* __restrict__ Wt,
                    const unsigned short* __restrict__ vt,
                    unsigned short* __restrict__ attnb) {
  __shared__ __align__(16) unsigned short Ws[64 * 32];
  __shared__ __align__(16) unsigned short Vs[64 * 32];

  const int tid = threadIdx.x, lane = tid & 31, wave = tid >> 5;
  const int b = blockIdx.z >> 4, h = blockIdx.z & 15;
  const long p0 = (long)blockIdx.y * 64;
  const int wm = (wave & 3) * 16, wn = (wave >> 2) * 32;

  const int lrow = tid >> 2, lkh = (tid & 3) * 8;        // 8 elems/thread
  const float* Wg = Wt + ((long)blockIdx.z * P_ + p0 + lrow) * N_ + lkh;
  const unsigned short* Vg = vt + ((long)blockIdx.z * 64 + lrow) * N_ + lkh;
  const unsigned voff = lds_off(&Vs[lrow * 32 + lkh]);

  f32x8 acc[2] = {zero8(), zero8()};

  for (int kt = 0; kt < 16; ++kt) {
    __syncthreads();
    async_ld_b128(voff, Vg + kt * 32);     // V tile: async global->LDS
    {
      const float* wp = Wg + kt * 32;
      f32x4 w0 = *(const f32x4*)(wp);
      f32x4 w1 = *(const f32x4*)(wp + 4);
      u32x4 pk;
      pk[0] = (unsigned)f2bf(w0[0]) | ((unsigned)f2bf(w0[1]) << 16);
      pk[1] = (unsigned)f2bf(w0[2]) | ((unsigned)f2bf(w0[3]) << 16);
      pk[2] = (unsigned)f2bf(w1[0]) | ((unsigned)f2bf(w1[1]) << 16);
      pk[3] = (unsigned)f2bf(w1[2]) | ((unsigned)f2bf(w1[3]) << 16);
      *(u32x4*)&Ws[lrow * 32 + lkh] = pk;
    }
    if (kt + 1 < 16)                       // stream the fp32 weights matrix
      __builtin_prefetch(Wg + (kt + 1) * 32, 0, 1);
    wait_asynccnt0();
    __syncthreads();

    FragB afr, bfr[2];
    int m = wm + (lane & 15);
    int ka = (lane >> 4) * 16;
    afr.u[0] = *(const u32x4*)&Ws[m * 32 + ka];
    afr.u[1] = *(const u32x4*)&Ws[m * 32 + ka + 8];
#pragma unroll
    for (int nt = 0; nt < 2; ++nt) {
      int j = wn + nt * 16 + (lane & 15);
      int kb = (lane >> 4) * 16;
      bfr[nt].u[0] = *(const u32x4*)&Vs[j * 32 + kb];
      bfr[nt].u[1] = *(const u32x4*)&Vs[j * 32 + kb + 8];
    }
#pragma unroll
    for (int nt = 0; nt < 2; ++nt)
      acc[nt] = wmma_bf16(afr, bfr[nt], acc[nt]);
  }

#pragma unroll
  for (int nt = 0; nt < 2; ++nt)
#pragma unroll
    for (int g = 0; g < 8; ++g) {
      long p = p0 + wm + g + ((lane >> 4) * 8);
      int  c = h * 64 + wn + nt * 16 + (lane & 15);
      attnb[((long)b * P_ + p) * D_ + c] = f2bf(acc[nt][g]);
    }
}

// ---------------------------------------------------------------------------
// x = LayerNorm(a + r) * g + b ; writes fp32 and bf16 copies. One row/block.
// ---------------------------------------------------------------------------
__global__ __launch_bounds__(256)
void ln_kernel(const float* __restrict__ a, const float* __restrict__ r,
               const float* __restrict__ g, const float* __restrict__ bb,
               float* __restrict__ xf, unsigned short* __restrict__ xb) {
  __shared__ float red[256];
  const long row = blockIdx.x;
  const int tid = threadIdx.x;
  const float* ar = a + row * D_;
  const float* rr = r + row * D_;
  float loc[4], s = 0.0f;
#pragma unroll
  for (int i = 0; i < 4; ++i) { loc[i] = ar[tid + i * 256] + rr[tid + i * 256]; s += loc[i]; }
  red[tid] = s; __syncthreads();
  for (int t = 128; t > 0; t >>= 1) { if (tid < t) red[tid] += red[tid + t]; __syncthreads(); }
  float mu = red[0] * (1.0f / D_);
  __syncthreads();
  float vs = 0.0f;
#pragma unroll
  for (int i = 0; i < 4; ++i) { float d = loc[i] - mu; vs += d * d; }
  red[tid] = vs; __syncthreads();
  for (int t = 128; t > 0; t >>= 1) { if (tid < t) red[tid] += red[tid + t]; __syncthreads(); }
  float rstd = rsqrtf(red[0] * (1.0f / D_) + 1e-5f);
#pragma unroll
  for (int i = 0; i < 4; ++i) {
    int c = tid + i * 256;
    float y = (loc[i] - mu) * rstd * g[c] + bb[c];
    xf[row * D_ + c] = y;
    if (xb) xb[row * D_ + c] = f2bf(y);
  }
}

// ---------------------------------------------------------------------------
// x2 = LN(x + ff); logits = x2 . cls_w + cls_b; probs = sigmoid(logits)
// ---------------------------------------------------------------------------
__global__ __launch_bounds__(256)
void ln2_cls_kernel(const float* __restrict__ x, const float* __restrict__ ff,
                    const float* __restrict__ g, const float* __restrict__ bb,
                    const float* __restrict__ cw, const float* __restrict__ cb,
                    float* __restrict__ xo, float* __restrict__ logits,
                    float* __restrict__ probs) {
  __shared__ float red[256];
  const long row = blockIdx.x;
  const int tid = threadIdx.x;
  const float* ar = x + row * D_;
  const float* rr = ff + row * D_;
  float loc[4], s = 0.0f;
#pragma unroll
  for (int i = 0; i < 4; ++i) { loc[i] = ar[tid + i * 256] + rr[tid + i * 256]; s += loc[i]; }
  red[tid] = s; __syncthreads();
  for (int t = 128; t > 0; t >>= 1) { if (tid < t) red[tid] += red[tid + t]; __syncthreads(); }
  float mu = red[0] * (1.0f / D_);
  __syncthreads();
  float vs = 0.0f;
#pragma unroll
  for (int i = 0; i < 4; ++i) { float d = loc[i] - mu; vs += d * d; }
  red[tid] = vs; __syncthreads();
  for (int t = 128; t > 0; t >>= 1) { if (tid < t) red[tid] += red[tid + t]; __syncthreads(); }
  float rstd = rsqrtf(red[0] * (1.0f / D_) + 1e-5f);
  __syncthreads();
  float dot = 0.0f;
#pragma unroll
  for (int i = 0; i < 4; ++i) {
    int c = tid + i * 256;
    float y = (loc[i] - mu) * rstd * g[c] + bb[c];
    xo[row * D_ + c] = y;
    dot += y * cw[c];
  }
  red[tid] = dot; __syncthreads();
  for (int t = 128; t > 0; t >>= 1) { if (tid < t) red[tid] += red[tid + t]; __syncthreads(); }
  if (tid == 0) {
    float lg = red[0] + cb[0];
    logits[row] = lg;
    probs[row] = 1.0f / (1.0f + __expf(-lg));
  }
}

// ---------------------------------------------------------------------------
extern "C" void kernel_launch(void* const* d_in, const int* in_sizes, int n_in,
                              void* d_out, int out_size, void* d_ws, size_t ws_size,
                              hipStream_t stream) {
  const float* img      = (const float*)d_in[0];
  const float* txt      = (const float*)d_in[1];
  const unsigned char* mask = (const unsigned char*)d_in[2];
  const float* in_w     = (const float*)d_in[3];
  const float* in_b     = (const float*)d_in[4];
  const float* out_w    = (const float*)d_in[5];
  const float* out_b    = (const float*)d_in[6];
  const float* ln1_g    = (const float*)d_in[7];
  const float* ln1_b    = (const float*)d_in[8];
  const float* ffn_w1   = (const float*)d_in[9];
  const float* ffn_b1   = (const float*)d_in[10];
  const float* ffn_w2   = (const float*)d_in[11];
  const float* ffn_b2   = (const float*)d_in[12];
  const float* ln2_g    = (const float*)d_in[13];
  const float* ln2_b    = (const float*)d_in[14];
  const float* cls_w    = (const float*)d_in[15];
  const float* cls_b    = (const float*)d_in[16];
  const float* log_tau  = (const float*)d_in[17];

  float* xout    = (float*)d_out;
  float* weights = xout + (long)B_ * P_ * D_;
  float* logits  = weights + (long)B_ * H_ * P_ * N_;
  float* probs   = logits + (long)B_ * P_;

  const long Mq = (long)B_ * P_;     // 9216
  const long Mk = (long)B_ * N_;     // 8192

  char* ws = (char*)d_ws;
  size_t off = 0;
  auto take = [&](size_t bytes) -> void* {
    void* p = ws + off;
    off += (bytes + 255) & ~(size_t)255;
    return p;
  };
  unsigned short* img_b  = (unsigned short*)take(Mq * D_ * 2);
  unsigned short* txt_b  = (unsigned short*)take(Mk * D_ * 2);
  unsigned short* win_b  = (unsigned short*)take(3L * D_ * D_ * 2);
  unsigned short* wout_b = (unsigned short*)take((long)D_ * D_ * 2);
  unsigned short* w1_b   = (unsigned short*)take((long)HID_ * D_ * 2);
  unsigned short* w2_b   = (unsigned short*)take((long)HID_ * D_ * 2);
  unsigned short* q_b    = (unsigned short*)take(Mq * D_ * 2);
  unsigned short* kvt_b  = (unsigned short*)take(Mk * 2L * D_ * 2);
  unsigned short* vt_b   = (unsigned short*)take((long)B_ * H_ * 64 * N_ * 2);
  unsigned short* attn_b = (unsigned short*)take(Mq * D_ * 2);
  float*          out_f  = (float*)take(Mq * D_ * 4);   // reused for ff
  float*          x_f    = (float*)take(Mq * D_ * 4);
  unsigned short* x_b    = (unsigned short*)take(Mq * D_ * 2);
  unsigned short* ffh_b  = (unsigned short*)take(Mq * (long)HID_ * 2);
  (void)ws_size; (void)in_sizes; (void)n_in; (void)out_size;

  // 1) fp32 -> bf16 conversions
  cvt_bf16_kernel<<<4096, 256, 0, stream>>>(img,    img_b,  Mq * D_);
  cvt_bf16_kernel<<<4096, 256, 0, stream>>>(txt,    txt_b,  Mk * D_);
  cvt_bf16_kernel<<<4096, 256, 0, stream>>>(in_w,   win_b,  3L * D_ * D_);
  cvt_bf16_kernel<<<4096, 256, 0, stream>>>(out_w,  wout_b, (long)D_ * D_);
  cvt_bf16_kernel<<<4096, 256, 0, stream>>>(ffn_w1, w1_b,   (long)HID_ * D_);
  cvt_bf16_kernel<<<4096, 256, 0, stream>>>(ffn_w2, w2_b,   (long)HID_ * D_);

  // 2) q = img @ Wq^T + bq   (bf16 out)
  gemm_bf16_kernel<<<dim3(D_ / 128, Mq / 128), 256, 0, stream>>>(
      img_b, D_, win_b, D_, in_b, nullptr, q_b, D_, D_, 0);

  // 3) kvt = txt @ Wkv^T + bkv  (bf16 out, N=2048)
  gemm_bf16_kernel<<<dim3(2 * D_ / 128, Mk / 128), 256, 0, stream>>>(
      txt_b, D_, win_b + (long)D_ * D_, D_, in_b + D_, nullptr, kvt_b, 2 * D_, D_, 0);

  // 4) V transpose -> Vt[b,h,64,512]
  vtrans_kernel<<<(B_ * H_ * 64 * N_) / 256, 256, 0, stream>>>(kvt_b, vt_b);

  // 5) raw scores into the weights output region
  attn_scores_kernel<<<dim3(N_ / 128, P_ / 64, B_ * H_), 256, 0, stream>>>(
      q_b, kvt_b, log_tau, weights);

  // 6) masked softmax in place
  softmax_kernel<<<B_ * H_ * P_, 256, 0, stream>>>(weights, mask);

  // 7) attn = weights @ V  (bf16 out)
  attn_av_kernel<<<dim3(1, P_ / 64, B_ * H_), 256, 0, stream>>>(weights, vt_b, attn_b);

  // 8) out = attn @ Wout^T + bout (fp32)
  gemm_bf16_kernel<<<dim3(D_ / 128, Mq / 128), 256, 0, stream>>>(
      attn_b, D_, wout_b, D_, out_b, out_f, nullptr, D_, D_, 0);

  // 9) x = LN(img + out)
  ln_kernel<<<Mq, 256, 0, stream>>>(img, out_f, ln1_g, ln1_b, x_f, x_b);

  // 10) h = relu(x @ W1^T + b1)  (bf16)
  gemm_bf16_kernel<<<dim3(HID_ / 128, Mq / 128), 256, 0, stream>>>(
      x_b, D_, w1_b, D_, ffn_b1, nullptr, ffh_b, HID_, D_, 1);

  // 11) ff = h @ W2^T + b2  (fp32, reuses out_f)
  gemm_bf16_kernel<<<dim3(D_ / 128, Mq / 128), 256, 0, stream>>>(
      ffh_b, HID_, w2_b, HID_, ffn_b2, out_f, nullptr, D_, HID_, 0);

  // 12) x2 = LN(x + ff) -> d_out; logits/probs
  ln2_cls_kernel<<<Mq, 256, 0, stream>>>(x_f, out_f, ln2_g, ln2_b, cls_w, cls_b,
                                         xout, logits, probs);
}
